// NTXentLoss_83391085019210
// MI455X (gfx1250) — matrix operand order repaired and checked by reading, not compiled, not run on hardware
//
#include <hip/hip_runtime.h>
#include <math.h>

typedef __attribute__((ext_vector_type(16))) _Float16 v16h;
typedef __attribute__((ext_vector_type(8)))  _Float16 v8h;
typedef __attribute__((ext_vector_type(8)))  float    v8f;
typedef __attribute__((ext_vector_type(4)))  float    v4f;

#define NB      4096          // B
#define TWOB    8192          // 2B
#define DDIM    256           // D
#define INV_T   (1.0f / 0.07f)

// ---------------------------------------------------------------------------
// Kernel 1: row-normalize z = [z_i ; z_j] -> f16, and zero the output scalar.
// One wave32 per row (8 floats per lane), 8 waves per block.
// ---------------------------------------------------------------------------
__global__ __launch_bounds__(256)
void ntxent_normalize(const float* __restrict__ zi,
                      const float* __restrict__ zj,
                      _Float16* __restrict__ zn,
                      float* __restrict__ out)
{
    const int wave = threadIdx.x >> 5;
    const int lane = threadIdx.x & 31;
    const int row  = blockIdx.x * 8 + wave;          // 0..8191

    const float* src = (row < NB) ? (zi + (size_t)row * DDIM)
                                  : (zj + (size_t)(row - NB) * DDIM);

    v4f x0 = *(const v4f*)(src + lane * 8);
    v4f x1 = *(const v4f*)(src + lane * 8 + 4);

    float ss = 0.0f;
#pragma unroll
    for (int i = 0; i < 4; ++i) ss += x0[i] * x0[i] + x1[i] * x1[i];
#pragma unroll
    for (int off = 16; off >= 1; off >>= 1)
        ss += __shfl_xor(ss, off, 32);

    float nrm = sqrtf(ss);
    nrm = fmaxf(nrm, 1e-8f);                          // COS_EPS clamp
    const float inv = 1.0f / nrm;

    v8h o;
#pragma unroll
    for (int i = 0; i < 4; ++i) {
        o[i]     = (_Float16)(x0[i] * inv);
        o[i + 4] = (_Float16)(x1[i] * inv);
    }
    *(v8h*)(zn + (size_t)row * DDIM + lane * 8) = o;

    if (blockIdx.x == 0 && threadIdx.x == 0) out[0] = 0.0f;
}

// ---------------------------------------------------------------------------
// Tile load helpers (layouts per CDNA5 ISA 7.12.2, wave32)
// ---------------------------------------------------------------------------

// A (16-bit 16x32): lane-half 0 holds K {32c..32c+7, 32c+16..32c+23}, half 1 is +8.
__device__ __forceinline__ void load_a_tile(const _Float16* __restrict__ zn,
                                            int Rrow, int half, v16h (&Av)[8])
{
    const _Float16* arow = zn + (size_t)Rrow * DDIM + half * 8;
#pragma unroll
    for (int c = 0; c < 8; ++c) {
        const v8h* p = (const v8h*)(arow + c * 32);
        v8h lo = p[0];            // K +0..7
        v8h hi = p[2];            // K +16..23
        Av[c] = __builtin_shufflevector(lo, hi,
                  0, 1, 2, 3, 4, 5, 6, 7, 8, 9, 10, 11, 12, 13, 14, 15);
    }
}

// B (32x16): lane = column, lane-half 0 holds K {32c..32c+7, 32c+16..32c+23}, half 1 is +8.
__device__ __forceinline__ void load_b_tile(const _Float16* __restrict__ zn,
                                            int Cb, int sub, int half, v16h (&Bv)[8])
{
    const _Float16* brow = zn + (size_t)(Cb + sub) * DDIM + half * 8;
#pragma unroll
    for (int c = 0; c < 8; ++c) {
        const v8h* p = (const v8h*)(brow + c * 32);
        v8h lo = p[0];            // K +0..7
        v8h hi = p[2];            // K +16..23
        Bv[c] = __builtin_shufflevector(lo, hi,
                  0, 1, 2, 3, 4, 5, 6, 7, 8, 9, 10, 11, 12, 13, 14, 15);
    }
}

// 16 WMMAs (2 row tiles x K=256) + fused masked-softmax epilogue.
__device__ __forceinline__ void compute_tile(const v16h (&A0)[8], const v16h (&A1)[8],
                                             const v16h (&Bv)[8],
                                             int Rb, int Cb, int sub, int half,
                                             float (&sumAcc)[16], float (&labAcc)[16])
{
    v8f acc0 = {};
    v8f acc1 = {};
#pragma unroll
    for (int c = 0; c < 8; ++c) {
        acc0 = __builtin_amdgcn_wmma_f32_16x16x32_f16(
                   false, A0[c], false, Bv[c], (short)0, acc0, false, false);
        acc1 = __builtin_amdgcn_wmma_f32_16x16x32_f16(
                   false, A1[c], false, Bv[c], (short)0, acc1, false, false);
    }
    const int ncol = Cb + sub;    // C layout: lane%16 = column
#pragma unroll
    for (int t = 0; t < 2; ++t) {
#pragma unroll
        for (int r = 0; r < 8; ++r) {
            const int   mg = Rb + t * 16 + r + half * 8;       // global row
            const float v  = t ? acc1[r] : acc0[r];
            const float e  = __expf((v - 1.0f) * INV_T);       // shift = 1/T
            sumAcc[t * 8 + r] += (ncol == mg) ? 0.0f : e;      // mask diagonal
            labAcc[t * 8 + r] += (ncol == ((mg + NB) & (TWOB - 1))) ? v : 0.0f;
        }
    }
}

// ---------------------------------------------------------------------------
// Kernel 2: fused  sim = zn @ zn^T  + masked softmax + NLL.
// Block = 32 rows (2 A tiles per wave); 8 waves = column groups (stride 8).
// B tiles are software-pipelined through ping-pong register buffers.
// ---------------------------------------------------------------------------
__global__ __launch_bounds__(256)
void ntxent_main(const _Float16* __restrict__ zn, float* __restrict__ out)
{
    __shared__ float s_sum[32];
    __shared__ float s_lab[32];

    const int lane = threadIdx.x & 31;
    const int wave = threadIdx.x >> 5;    // 0..7 = column group
    const int half = lane >> 4;           // 0 or 1
    const int sub  = lane & 15;           // row (A) / col (B,C) within tile
    const int Rb   = blockIdx.x * 32;     // 32-row block base

    if (threadIdx.x < 32) { s_sum[threadIdx.x] = 0.0f; s_lab[threadIdx.x] = 0.0f; }
    __syncthreads();

    v16h A0[8], A1[8];
    load_a_tile(zn, Rb + sub,      half, A0);
    load_a_tile(zn, Rb + 16 + sub, half, A1);

    float sumAcc[16], labAcc[16];
#pragma unroll
    for (int i = 0; i < 16; ++i) { sumAcc[i] = 0.0f; labAcc[i] = 0.0f; }

    // Software pipeline: wave handles tiles j = wave + 8k, k = 0..63 (pairs).
    v16h B0[8], B1[8];
    load_b_tile(zn, wave * 16, sub, half, B0);

    for (int pair = 0; pair < 32; ++pair) {
        const int j0 = wave + pair * 16;                      // even-k tile
        load_b_tile(zn, (j0 + 8) * 16, sub, half, B1);        // prefetch odd-k
        compute_tile(A0, A1, B0, Rb, j0 * 16, sub, half, sumAcc, labAcc);
        if (pair < 31)
            load_b_tile(zn, (j0 + 16) * 16, sub, half, B0);   // prefetch next even
        compute_tile(A0, A1, B1, Rb, (j0 + 8) * 16, sub, half, sumAcc, labAcc);
    }

    // ---- combine 16 lanes x 8 waves per row via LDS float atomics ----
#pragma unroll
    for (int t = 0; t < 2; ++t) {
#pragma unroll
        for (int r = 0; r < 8; ++r) {
            const int ml = t * 16 + r + half * 8;
            atomicAdd(&s_sum[ml], sumAcc[t * 8 + r]);
            atomicAdd(&s_lab[ml], labAcc[t * 8 + r]);
        }
    }
    __syncthreads();

    if (threadIdx.x < 32) {
        const float sum = s_sum[threadIdx.x];
        const float lab = s_lab[threadIdx.x];
        // -logp = (1/T + log(sum)) - lab/T
        const float loss = (1.0f - lab) * INV_T + __logf(sum);
        atomicAdd(out, loss * (1.0f / (float)TWOB));
    }
}

// ---------------------------------------------------------------------------
extern "C" void kernel_launch(void* const* d_in, const int* in_sizes, int n_in,
                              void* d_out, int out_size, void* d_ws, size_t ws_size,
                              hipStream_t stream)
{
    const float* zi = (const float*)d_in[0];
    const float* zj = (const float*)d_in[1];
    float* out = (float*)d_out;
    _Float16* zn = (_Float16*)d_ws;   // 8192 x 256 f16 = 4 MB scratch

    ntxent_normalize<<<TWOB / 8, 256, 0, stream>>>(zi, zj, zn, out);
    ntxent_main<<<TWOB / 32, 256, 0, stream>>>(zn, out);
}
